// Block_57801669870145
// MI455X (gfx1250) — matrix operand order repaired and checked by reading, not compiled, not run on hardware
//
#include <hip/hip_runtime.h>
#include <hip/hip_bf16.h>
#include <math.h>
#include <stdint.h>

// ---------- types ----------
typedef __attribute__((ext_vector_type(16))) __bf16 v16bf;
typedef __attribute__((ext_vector_type(8)))  __bf16 v8bf;
typedef __attribute__((ext_vector_type(8)))  float  v8f;
typedef __attribute__((ext_vector_type(4)))  int    v4i;

#define GN    50000
#define GE    800000
#define GFIN  128
#define GFOUT 256
#define GHID  32
#define GEPS  1e-5f

// ---- gfx1250 async global->LDS copy (guarded; falls back to sync copy) ----
#ifdef __has_builtin
#if __has_builtin(__builtin_amdgcn_global_load_async_to_lds_b128)
#define HAVE_ASYNC_LDS 1
#endif
#if __has_builtin(__builtin_amdgcn_s_wait_asynccnt)
#define HAVE_WAIT_ASYNC 1
#endif
#endif
#ifndef HAVE_ASYNC_LDS
#define HAVE_ASYNC_LDS 0
#endif
#ifndef HAVE_WAIT_ASYNC
#define HAVE_WAIT_ASYNC 0
#endif

#if HAVE_ASYNC_LDS
typedef __attribute__((address_space(1))) v4i as1_v4i;  // global int4
typedef __attribute__((address_space(3))) v4i as3_v4i;  // LDS int4
#endif

static __device__ __forceinline__ void async_copy16(const void* g, void* l) {
#if HAVE_ASYNC_LDS
  // AS1 global source, AS3 LDS dest; flat LDS addr low 32 bits == LDS offset.
  __builtin_amdgcn_global_load_async_to_lds_b128(
      (as1_v4i*)(uintptr_t)g, (as3_v4i*)(uintptr_t)l, 0, 0);
#else
  *(uint4*)l = *(const uint4*)g;
#endif
}

static __device__ __forceinline__ void async_wait() {
#if HAVE_ASYNC_LDS
#if HAVE_WAIT_ASYNC
  __builtin_amdgcn_s_wait_asynccnt(0);
#else
  asm volatile("s_wait_asynccnt 0x0" ::: "memory");
#endif
#endif
}

static __device__ __forceinline__ v8f vzero8() {
  v8f z = {0.f,0.f,0.f,0.f,0.f,0.f,0.f,0.f};
  return z;
}

static __device__ __forceinline__ v16bf make_afrag(const __bf16* p_lo, const __bf16* p_hi) {
  v8bf lo = *(const v8bf*)p_lo;
  v8bf hi = *(const v8bf*)p_hi;
  v16bf a;
#pragma unroll
  for (int i = 0; i < 8; ++i) { a[i] = lo[i]; a[i + 8] = hi[i]; }
  return a;
}

// =====================================================================
// Kernel 1: edge MLP.  Per wave: one tile of 16 edges.
//   xd = |x[row]-x[col]| (16x128 bf16 in LDS);  h = relu(xd@w1+b1) via
//   4 K-steps x 2 n-tiles x 2 MLPs of v_wmma_f32_16x16x32_bf16;
//   v = sigmoid(h@w2+b2) via half-wave shuffle reduction.
// =====================================================================
__global__ void __launch_bounds__(256)
edge_mlp_kernel(const float* __restrict__ x,
                const int* __restrict__ row, const int* __restrict__ col,
                const float* __restrict__ w1a, const float* __restrict__ b1a,
                const float* __restrict__ w2a, const float* __restrict__ b2a,
                const float* __restrict__ w1b, const float* __restrict__ b1b,
                const float* __restrict__ w2b, const float* __restrict__ b2b,
                float* __restrict__ v1, float* __restrict__ v2) {
  __shared__ __bf16 sW1[2][GHID][GFIN];   // transposed weights [mlp][n][k], 16 KB
  __shared__ __bf16 sXd[8][16][GFIN];     // per-wave edge-diff tiles, 32 KB

  const int t    = threadIdx.x;
  const int lane = t & 31;
  const int wave = t >> 5;

  // stage w1 (both MLPs) transposed to [n][k] so B-fragments are contiguous
#pragma unroll 4
  for (int i = 0; i < 32; ++i) {
    int idx = i * 256 + t;            // 0..8191
    int mlp = idx >> 12;
    int rem = idx & 4095;
    int n = rem >> 7, k = rem & 127;
    const float* w = mlp ? w1b : w1a;
    sW1[mlp][n][k] = (__bf16)w[k * GHID + n];
  }
  __syncthreads();

  const int tile = blockIdx.x * 8 + wave;     // 50000 tiles total, exact
  const int eLoc = lane & 15;                 // edge within tile == matrix row M
  const int half = lane >> 4;
  const int e    = tile * 16 + eLoc;
  const int ra   = row[e];
  const int ca   = col[e];
  const float* xr = x + (size_t)ra * GFIN + half * 64;
  const float* xc = x + (size_t)ca * GFIN + half * 64;
  __builtin_prefetch(xr, 0, 0);
  __builtin_prefetch(xc, 0, 0);

  // stage |x[row]-x[col]| -> bf16 LDS (each lane: 64 columns)
#pragma unroll 8
  for (int i = 0; i < 64; ++i) {
    sXd[wave][eLoc][half * 64 + i] = (__bf16)fabsf(xr[i] - xc[i]);
  }
  __syncthreads();

  v8f acc[2][2];
#pragma unroll
  for (int m = 0; m < 2; ++m)
#pragma unroll
    for (int nt = 0; nt < 2; ++nt) acc[m][nt] = vzero8();

  const int aoff = (lane < 16) ? 0 : 16;  // A 16x32 bf16 layout: K half
  const int boff = (lane < 16) ? 0 : 16;  // B 32x16 bf16 layout: K half

#pragma unroll
  for (int kc = 0; kc < 4; ++kc) {
    v16bf a = make_afrag(&sXd[wave][eLoc][kc * 32 + aoff],
                         &sXd[wave][eLoc][kc * 32 + aoff + 8]);
#pragma unroll
    for (int mlp = 0; mlp < 2; ++mlp) {
#pragma unroll
      for (int nt = 0; nt < 2; ++nt) {
        v16bf b = *(const v16bf*)&sW1[mlp][nt * 16 + eLoc][kc * 32 + boff];
        acc[mlp][nt] = __builtin_amdgcn_wmma_f32_16x16x32_bf16(
            false, a, false, b, (short)0, acc[mlp][nt], false, false);
      }
    }
  }

  // second layer: relu(h + b1) . w2, reduce across the 16 lanes of each half
#pragma unroll
  for (int mlp = 0; mlp < 2; ++mlp) {
    const float* b1p = mlp ? b1b : b1a;
    const float* w2p = mlp ? w2b : w2a;
    const float  b2v = mlp ? b2b[0] : b2a[0];
    float* vout = mlp ? v2 : v1;
    float bLo = b1p[eLoc],      bHi = b1p[16 + eLoc];
    float wLo = w2p[eLoc],      wHi = w2p[16 + eLoc];
#pragma unroll
    for (int r = 0; r < 8; ++r) {
      float s = fmaxf(acc[mlp][0][r] + bLo, 0.f) * wLo
              + fmaxf(acc[mlp][1][r] + bHi, 0.f) * wHi;
      s += __shfl_xor(s, 1, 16);
      s += __shfl_xor(s, 2, 16);
      s += __shfl_xor(s, 4, 16);
      s += __shfl_xor(s, 8, 16);
      if (eLoc == 0) {
        int eo = tile * 16 + half * 8 + r;   // lanes<16 -> M=r, lanes>=16 -> M=8+r
        vout[eo] = 1.f / (1.f + __expf(-(s + b2v)));
      }
    }
  }
}

// =====================================================================
// Kernel 2: edge-parallel scatter SPMM: out_k[row] += v_k[e] * feats[col]
// (f32 atomics; x/h and targets are L2-resident)
// =====================================================================
__global__ void __launch_bounds__(256)
spmm_scatter_kernel(const int* __restrict__ row, const int* __restrict__ col,
                    const float* __restrict__ v1, const float* __restrict__ v2,
                    const float* __restrict__ feats,
                    float* __restrict__ o1, float* __restrict__ o2,
                    int E, int F, int perLog2) {
  long long tid = (long long)blockIdx.x * blockDim.x + threadIdx.x;
  long long e = tid >> perLog2;
  if (e >= E) return;
  int cq = (int)(tid & ((1 << perLog2) - 1)) << 2;
  int r = row[e], c = col[e];
  float a = v1[e], b = v2[e];
  const float4 f = *(const float4*)(feats + (size_t)c * F + cq);
  float* p1 = o1 + (size_t)r * F + cq;
  float* p2 = o2 + (size_t)r * F + cq;
  atomicAdd(p1 + 0, a * f.x); atomicAdd(p1 + 1, a * f.y);
  atomicAdd(p1 + 2, a * f.z); atomicAdd(p1 + 3, a * f.w);
  atomicAdd(p2 + 0, b * f.x); atomicAdd(p2 + 1, b * f.y);
  atomicAdd(p2 + 2, b * f.z); atomicAdd(p2 + 3, b * f.w);
}

// =====================================================================
// Kernel 3a: pack the virtual concat [S0*rowscale | S1 | S2] -> bf16 [M][Ktot]
// =====================================================================
__global__ void __launch_bounds__(128)
pack_a_kernel(const float* __restrict__ S0, const float* __restrict__ S1,
              const float* __restrict__ S2, const float* __restrict__ rowscale,
              int Ks, int ksLog2, __bf16* __restrict__ out, int Ktot) {
  int m = blockIdx.y;
  int k = blockIdx.x * 128 + threadIdx.x;
  int src = k >> ksLog2;
  int kl  = k & (Ks - 1);
  const float* sp = (src == 0) ? S0 : ((src == 1) ? S1 : S2);
  float v = sp[(size_t)m * Ks + kl];
  if (src == 0 && rowscale) v *= rowscale[m];
  out[(size_t)m * Ktot + k] = (__bf16)v;
}

// =====================================================================
// Kernel 3b: transpose + convert weights: W[Ktot][256] -> WT bf16 [256][Ktot]
// =====================================================================
__global__ void __launch_bounds__(256)
pack_w_kernel(const float* __restrict__ W, __bf16* __restrict__ out, int Ktot) {
  int n = blockIdx.x;   // 0..255
  for (int k = threadIdx.x; k < Ktot; k += blockDim.x)
    out[(size_t)n * Ktot + k] = (__bf16)W[(size_t)k * GFOUT + n];
}

// =====================================================================
// Kernel 4: packed-bf16 WMMA GEMM:  Z[M][256] = A[M][Ktot] @ WT^T + bias
//  - 32-row A tile async-DMA'd global->LDS once per block (ASYNCcnt path)
//  - B fragments read straight from L2 (WT is [n][k]-contiguous, 32B/frag)
//  - optional per-channel sum / sum-of-squares for BatchNorm
// =====================================================================
__global__ void __launch_bounds__(256)
gemm_wmma_kernel(const __bf16* __restrict__ A, const __bf16* __restrict__ WT,
                 const float* __restrict__ bias,
                 float* __restrict__ Z, float* __restrict__ stats,
                 int Ktot, int M) {
  __shared__ __bf16 sA[32 * 768];   // up to 48 KB A tile (32 rows x Ktot)

  const int t     = threadIdx.x;
  const int lane  = t & 31;
  const int wave  = t >> 5;
  const int mBase = blockIdx.x * 32;

  // ---- stage A tile: one contiguous (32*Ktot*2)-byte async copy ----
  {
    const int totalChunks = 4 * Ktot;                 // 16B chunks
    const size_t tileByte = (size_t)mBase * Ktot * 2; // byte offset into A
    const size_t maxByte  = (size_t)M * Ktot * 2 - 16;
    for (int c = t; c < totalChunks; c += 256) {
      size_t gb = tileByte + (size_t)c * 16;
      if (gb > maxByte) gb = maxByte;                 // clamp tail block
      async_copy16((const char*)A + gb, (char*)sA + (size_t)c * 16);
    }
    async_wait();
  }
  __syncthreads();

  const int mt   = wave >> 2;         // 0..1  (16-row sub-tile)
  const int nq   = wave & 3;          // 0..3  (4 n-tiles each)
  const int mrow = lane & 15;
  const int aoff = (lane < 16) ? 0 : 16;
  const int boff = (lane < 16) ? 0 : 16;
  const __bf16* aRow = &sA[(size_t)(mt * 16 + mrow) * Ktot];

  v8f acc[4];
#pragma unroll
  for (int j = 0; j < 4; ++j) acc[j] = vzero8();

  for (int k0 = 0; k0 < Ktot; k0 += 32) {
    v16bf a = make_afrag(aRow + k0 + aoff, aRow + k0 + aoff + 8);
#pragma unroll
    for (int j = 0; j < 4; ++j) {
      int n = (nq * 4 + j) * 16 + mrow;
      v16bf b = *(const v16bf*)(WT + (size_t)n * Ktot + k0 + boff);
      acc[j] = __builtin_amdgcn_wmma_f32_16x16x32_bf16(
          false, a, false, b, (short)0, acc[j], false, false);
    }
  }

  // ---- epilogue: bias, store, batch-stat accumulation ----
#pragma unroll
  for (int j = 0; j < 4; ++j) {
    int n = (nq * 4 + j) * 16 + mrow;
    float bv = bias ? bias[n] : 0.f;
    float s = 0.f, ss = 0.f;
#pragma unroll
    for (int r = 0; r < 8; ++r) {
      int mg = mBase + mt * 16 + ((lane < 16) ? r : (8 + r));
      if (mg < M) {
        float z = acc[j][r] + bv;
        Z[(size_t)mg * GFOUT + n] = z;
        s += z; ss += z * z;
      }
    }
    if (stats) {
      s  += __shfl_xor(s, 16);
      ss += __shfl_xor(ss, 16);
      if (lane < 16) {
        atomicAdd(&stats[n], s);
        atomicAdd(&stats[GFOUT + n], ss);
      }
    }
  }
}

// =====================================================================
// Kernel 5: BatchNorm apply (+optional residual) + ReLU
// =====================================================================
__global__ void __launch_bounds__(256)
bn_relu_kernel(const float* __restrict__ z, const float* __restrict__ stats,
               const float* __restrict__ g, const float* __restrict__ bta,
               const float* __restrict__ resid, float* __restrict__ out, int M) {
  size_t idx = (size_t)blockIdx.x * blockDim.x + threadIdx.x;
  if (idx >= (size_t)M * GFOUT) return;
  int n = (int)(idx & (GFOUT - 1));
  float invM = 1.f / (float)M;
  float mu  = stats[n] * invM;
  float var = stats[GFOUT + n] * invM - mu * mu;
  float sc  = rsqrtf(var + GEPS) * g[n];
  float val = (z[idx] - mu) * sc + bta[n];
  if (resid) val += resid[idx];
  out[idx] = fmaxf(val, 0.f);
}

// =====================================================================
// Launch
// =====================================================================
extern "C" void kernel_launch(void* const* d_in, const int* in_sizes, int n_in,
                              void* d_out, int out_size, void* d_ws, size_t ws_size,
                              hipStream_t stream) {
  (void)in_sizes; (void)n_in; (void)out_size; (void)ws_size;

  const float* x     = (const float*)d_in[0];
  const float* wid   = (const float*)d_in[1];
  const int*   widx  = (const int*)d_in[2];   // [2, E]
  // d_in[3] = win_values: overwritten by edge MLP in reference -> unused
  const float* m1w1 = (const float*)d_in[4];
  const float* m1b1 = (const float*)d_in[5];
  const float* m1w2 = (const float*)d_in[6];
  const float* m1b2 = (const float*)d_in[7];
  const float* m2w1 = (const float*)d_in[8];
  const float* m2b1 = (const float*)d_in[9];
  const float* m2w2 = (const float*)d_in[10];
  const float* m2b2 = (const float*)d_in[11];
  const float* fc1w = (const float*)d_in[12];
  const float* fc1b = (const float*)d_in[13];
  const float* bn1g = (const float*)d_in[14];
  const float* bn1b = (const float*)d_in[15];
  const float* fc2w = (const float*)d_in[16];
  const float* fc2b = (const float*)d_in[17];
  const float* bn2g = (const float*)d_in[18];
  const float* bn2b = (const float*)d_in[19];
  const float* resw = (const float*)d_in[20];

  const int* row = widx;
  const int* col = widx + GE;

  // ---- workspace layout ----
  float* ws = (float*)d_ws;
  size_t off = 0;
  float* a1    = ws + off; off += (size_t)GN * GFIN;    // spmm(v1, x)
  float* a2    = ws + off; off += (size_t)GN * GFIN;    // spmm(v2, x)
  float* b1    = ws + off; off += (size_t)GN * GFOUT;   // spmm(v1, h)
  float* b2    = ws + off; off += (size_t)GN * GFOUT;   // spmm(v2, h)
  float* stats = ws + off; off += 4 * GFOUT;            // [sum1|sq1|sum2|sq2]
  size_t zeroFloats = off;                              // atomic targets
  float* v1    = ws + off; off += GE;
  float* v2    = ws + off; off += GE;
  float* zbuf  = ws + off; off += (size_t)GN * GFOUT;
  float* hbuf  = ws + off; off += (size_t)GN * GFOUT;
  float* rbuf  = ws + off; off += (size_t)GN * GFOUT;
  __bf16* bfa = (__bf16*)(ws + off);
  size_t boff2 = 0;
  __bf16* Abf  = bfa + boff2; boff2 += (size_t)GN * 768;  // reused: K=384 then 768
  __bf16* xbf  = bfa + boff2; boff2 += (size_t)GN * GFIN;
  __bf16* wt1  = bfa + boff2; boff2 += (size_t)384 * GFOUT;
  __bf16* wt2  = bfa + boff2; boff2 += (size_t)768 * GFOUT;
  __bf16* wtr  = bfa + boff2; boff2 += (size_t)GFIN * GFOUT;

  (void)hipMemsetAsync(ws, 0, zeroFloats * sizeof(float), stream);

  // 0) pack weights (bf16, [n][k]) and x (bf16)
  pack_w_kernel<<<GFOUT, 256, 0, stream>>>(fc1w, wt1, 384);
  pack_w_kernel<<<GFOUT, 256, 0, stream>>>(fc2w, wt2, 768);
  pack_w_kernel<<<GFOUT, 256, 0, stream>>>(resw, wtr, GFIN);
  pack_a_kernel<<<dim3(GFIN / 128, GN), 128, 0, stream>>>(
      x, nullptr, nullptr, nullptr, GFIN, 7, xbf, GFIN);

  // 1) edge MLPs -> v1, v2   (50000 tiles of 16 edges, 8 waves/block)
  edge_mlp_kernel<<<GE / 16 / 8, 256, 0, stream>>>(
      x, row, col, m1w1, m1b1, m1w2, m1b2, m2w1, m2b1, m2w2, m2b2, v1, v2);

  // 2) spmm on x -> a1, a2
  {
    long long th = (long long)GE * (GFIN / 4);
    spmm_scatter_kernel<<<(unsigned)((th + 255) / 256), 256, 0, stream>>>(
        row, col, v1, v2, x, a1, a2, GE, GFIN, 5);
  }

  // 3) gconv1: pack A = [wid*x | a1 | a2], GEMM, stats -> stats[0..511]
  pack_a_kernel<<<dim3(384 / 128, GN), 128, 0, stream>>>(
      x, a1, a2, wid, GFIN, 7, Abf, 384);
  gemm_wmma_kernel<<<(GN + 31) / 32, 256, 0, stream>>>(
      Abf, wt1, fc1b, zbuf, stats, 384, GN);

  // 4) h = relu(bn1(z))
  bn_relu_kernel<<<(unsigned)(((size_t)GN * GFOUT + 255) / 256), 256, 0, stream>>>(
      zbuf, stats, bn1g, bn1b, nullptr, hbuf, GN);

  // 5) spmm on h -> b1, b2
  {
    long long th = (long long)GE * (GFOUT / 4);
    spmm_scatter_kernel<<<(unsigned)((th + 255) / 256), 256, 0, stream>>>(
        row, col, v1, v2, hbuf, b1, b2, GE, GFOUT, 6);
  }

  // 6) residual: r = x @ res_w
  gemm_wmma_kernel<<<(GN + 31) / 32, 256, 0, stream>>>(
      xbf, wtr, nullptr, rbuf, nullptr, GFIN, GN);

  // 7) gconv2: pack A = [wid*h | b1 | b2], GEMM, stats -> stats[512..1023]
  pack_a_kernel<<<dim3(768 / 128, GN), 128, 0, stream>>>(
      hbuf, b1, b2, wid, GFOUT, 8, Abf, 768);
  gemm_wmma_kernel<<<(GN + 31) / 32, 256, 0, stream>>>(
      Abf, wt2, fc2b, zbuf, stats + 2 * GFOUT, 768, GN);

  // 8) out = relu(bn2(z) + r)
  bn_relu_kernel<<<(unsigned)(((size_t)GN * GFOUT + 255) / 256), 256, 0, stream>>>(
      zbuf, stats + 2 * GFOUT, bn2g, bn2b, rbuf, (float*)d_out, GN);
}
